// whole_network_33801392620001
// MI455X (gfx1250) — matrix operand order repaired and checked by reading, not compiled, not run on hardware
//
#include <hip/hip_runtime.h>

// ---------------------------------------------------------------------------
// MI455X (gfx1250) fused GAT network, round 3.
//  * v_wmma_f32_16x16x32_f16, wave tile 32x32 (4 accumulators), block 64x128
//  * B operands pre-transposed once in HBM; fragments = two 16B global
//    loads (L2-resident), zero LDS traffic for B
//  * A tiles: CDNA5 async copies (global_load_async_to_lds_b128 / ASYNCcnt),
//    branch-free software pipeline over two fixed LDS buffers
//  * compile-time K/N -> immediate-offset addressing in the hot loop
//  * attention matrices rank-1 structured: synthesized on the fly; only the
//    mandatory fc2 output touches HBM at [B,N,N] size
// ---------------------------------------------------------------------------

typedef _Float16 half_t;
typedef __attribute__((ext_vector_type(16))) _Float16 v16h;
typedef __attribute__((ext_vector_type(8)))  float    v8f;

union FragH { v16h v; uint4 q[2]; };

#define LDW 40  // LDS A-tile pitch in halves (32 payload + 8 pad) = 80B, 16B-aligned

__device__ __forceinline__ float leaky01(float x) { return x >= 0.f ? x : 0.01f * x; }

// CDNA5 async copy: global -> LDS, 16 bytes per lane, tracked on ASYNCcnt.
__device__ __forceinline__ void async_ld_b128(const half_t* gsrc, half_t* ldst) {
  unsigned loff = (unsigned)(size_t)ldst;  // LDS aperture: low 32 bits = LDS offset
  asm volatile("global_load_async_to_lds_b128 %0, %1, off"
               :: "v"(loff), "v"(gsrc) : "memory");
}
__device__ __forceinline__ void wait_async0() {
  asm volatile("s_wait_asynccnt 0" ::: "memory");
}

#define WMMA(A, B, C) \
  __builtin_amdgcn_wmma_f32_16x16x32_f16(false, (A).v, false, (B).v, (short)0, (C), false, false)

// ---------------------------------------------------------------------------
__global__ __launch_bounds__(256) void cvt4_f32_f16(const float* __restrict__ in,
                                                    half_t* __restrict__ out, int n4) {
  int i = blockIdx.x * 256 + threadIdx.x;
  if (i >= n4) return;
  float4 v = ((const float4*)in)[i];
  union { uint2 u; half_t h[4]; } r;
  r.h[0] = (half_t)v.x; r.h[1] = (half_t)v.y;
  r.h[2] = (half_t)v.z; r.h[3] = (half_t)v.w;
  ((uint2*)out)[i] = r.u;
}

// Tiled f16 transpose: in [b][R][C] -> out [b][C][R]
__global__ __launch_bounds__(256) void transpose16(const half_t* __restrict__ in,
                                                   half_t* __restrict__ out, int R, int C) {
  __shared__ half_t tile[32][33];
  int b = blockIdx.z;
  const half_t* ib = in + (size_t)b * R * C;
  half_t* ob = out + (size_t)b * R * C;
  int c0 = blockIdx.x * 32, r0 = blockIdx.y * 32;
  int tx = threadIdx.x & 31, ty = threadIdx.x >> 5;
#pragma unroll
  for (int i = 0; i < 4; ++i)
    tile[ty + 8 * i][tx] = ib[(size_t)(r0 + ty + 8 * i) * C + c0 + tx];
  __syncthreads();
#pragma unroll
  for (int i = 0; i < 4; ++i)
    ob[(size_t)(c0 + ty + 8 * i) * R + r0 + tx] = tile[tx][ty + 8 * i];
}

// ---------------------------------------------------------------------------
// C[M,NN] = A[M,K] @ Bt[NN,K]^T  (f16 in, f32 acc, f16 out)
// Block 64x128, 8 waves = 2(M) x 4(N), wave tile 32x32, BK=32.
// ---------------------------------------------------------------------------
template <int K, int NN>
__global__ __launch_bounds__(256) void gemm_bt(const half_t* __restrict__ A,
                                               const half_t* __restrict__ Bt,
                                               half_t* __restrict__ C) {
  __shared__ half_t As[2][64 * LDW];
  const int t = threadIdx.x;
  const int m0 = blockIdx.y * 64;
  const int n0 = blockIdx.x * 128;
  const int lane = t & 31, w = t >> 5;
  const int wn = w & 3, wm = w >> 2;
  const int g = lane >> 4, l16 = lane & 15;
  const int srow = t >> 2, sc4 = t & 3;  // staging coords: 8 halves/thread
  const half_t* asrc = A + (size_t)(m0 + srow) * K + sc4 * 8;
  half_t* adst0 = &As[0][srow * LDW + sc4 * 8];
  half_t* adst1 = &As[1][srow * LDW + sc4 * 8];
  const half_t* brow0 = Bt + (size_t)(n0 + wn * 32 + l16) * K + g * 8;
  const half_t* brow1 = brow0 + (size_t)16 * K;
  const int afo = (wm * 32 + l16) * LDW + g * 8;  // A-frag base offset in tile

  v8f acc00 = {}, acc01 = {}, acc10 = {}, acc11 = {};

  auto compute = [&](int buf, int k0) {
    FragH a0, a1, b0, b1;
    const half_t* Ap = &As[buf][afo];
    a0.q[0] = *(const uint4*)Ap;                 a0.q[1] = *(const uint4*)(Ap + 16);
    a1.q[0] = *(const uint4*)(Ap + 16 * LDW);    a1.q[1] = *(const uint4*)(Ap + 16 * LDW + 16);
    b0.q[0] = *(const uint4*)(brow0 + k0);       b0.q[1] = *(const uint4*)(brow0 + k0 + 16);
    b1.q[0] = *(const uint4*)(brow1 + k0);       b1.q[1] = *(const uint4*)(brow1 + k0 + 16);
    acc00 = WMMA(a0, b0, acc00);
    acc01 = WMMA(a0, b1, acc01);
    acc10 = WMMA(a1, b0, acc10);
    acc11 = WMMA(a1, b1, acc11);
  };

  // branch-free double-buffered pipeline (K/32 is even for all uses)
  async_ld_b128(asrc, adst0);
  wait_async0();
  __syncthreads();
  int k0 = 0;
#pragma unroll 1
  for (; k0 < K - 64; k0 += 64) {
    async_ld_b128(asrc + k0 + 32, adst1);
    compute(0, k0);
    wait_async0();
    __syncthreads();
    async_ld_b128(asrc + k0 + 64, adst0);
    compute(1, k0 + 32);
    wait_async0();
    __syncthreads();
  }
  async_ld_b128(asrc + k0 + 32, adst1);
  compute(0, k0);
  wait_async0();
  __syncthreads();
  compute(1, k0 + 32);

#pragma unroll
  for (int ms = 0; ms < 2; ++ms)
#pragma unroll
    for (int ns = 0; ns < 2; ++ns) {
      const v8f& cc = ms ? (ns ? acc11 : acc10) : (ns ? acc01 : acc00);
      int gc = n0 + wn * 32 + ns * 16 + l16;
#pragma unroll
      for (int r = 0; r < 8; ++r) {
        int gr = m0 + wm * 32 + ms * 16 + g * 8 + r;
        C[(size_t)gr * NN + gc] = (half_t)cc[r];
      }
    }
}

// ---------------------------------------------------------------------------
// Fused attention GEMM: out[b] = act( att_b @ H_b ), K = 1024 nodes.
// att(i,j) = sigmoid(leaky(el_i+er_j)*scale_b + off_b), synthesized per step
// (8 sigmoids/thread) into double-buffered LDS.  B = Ht[b][FO][1024].
// ---------------------------------------------------------------------------
template <bool ELU, bool OUT16, int FO>
__global__ __launch_bounds__(256) void gat_att_gemm(const float* __restrict__ el,
                                                    const float* __restrict__ er,
                                                    const float2* __restrict__ sc,
                                                    const half_t* __restrict__ Ht,
                                                    half_t* __restrict__ C16,
                                                    float* __restrict__ C32) {
  __shared__ half_t As[2][64 * LDW];
  const int b = blockIdx.z;
  const float2 so = sc[b];
  const float* erb = er + (b << 10);
  const half_t* Bt = Ht + (size_t)b * 1024 * FO;  // [FO][1024]
  const int t = threadIdx.x;
  const int m0 = blockIdx.y * 64;
  const int n0 = blockIdx.x * 128;
  const int lane = t & 31, w = t >> 5;
  const int wn = w & 3, wm = w >> 2;
  const int g = lane >> 4, l16 = lane & 15;
  const int srow = t >> 2, sc4 = t & 3;
  const float ei = el[(b << 10) + m0 + srow];  // loop-invariant
  const half_t* brow0 = Bt + (size_t)(n0 + wn * 32 + l16) * 1024 + g * 8;
  const half_t* brow1 = brow0 + (size_t)16 * 1024;
  const int afo = (wm * 32 + l16) * LDW + g * 8;
  const int sso = srow * LDW + sc4 * 8;

  auto synth = [&](int buf, int k0) {
    float ev[8];
    *(float4*)&ev[0] = *(const float4*)&erb[k0 + sc4 * 8];
    *(float4*)&ev[4] = *(const float4*)&erb[k0 + sc4 * 8 + 4];
    union { uint4 q; half_t h[8]; } av;
#pragma unroll
    for (int e = 0; e < 8; ++e) {
      float y = leaky01(ei + ev[e]) * so.x + so.y;
      av.h[e] = (half_t)(1.f / (1.f + __expf(-y)));
    }
    *(uint4*)&As[buf][sso] = av.q;
  };

  v8f acc00 = {}, acc01 = {}, acc10 = {}, acc11 = {};

  auto compute = [&](int buf, int k0) {
    FragH a0, a1, b0, b1;
    const half_t* Ap = &As[buf][afo];
    a0.q[0] = *(const uint4*)Ap;                 a0.q[1] = *(const uint4*)(Ap + 16);
    a1.q[0] = *(const uint4*)(Ap + 16 * LDW);    a1.q[1] = *(const uint4*)(Ap + 16 * LDW + 16);
    b0.q[0] = *(const uint4*)(brow0 + k0);       b0.q[1] = *(const uint4*)(brow0 + k0 + 16);
    b1.q[0] = *(const uint4*)(brow1 + k0);       b1.q[1] = *(const uint4*)(brow1 + k0 + 16);
    acc00 = WMMA(a0, b0, acc00);
    acc01 = WMMA(a0, b1, acc01);
    acc10 = WMMA(a1, b0, acc10);
    acc11 = WMMA(a1, b1, acc11);
  };

  synth(0, 0);
  __syncthreads();
  int k0 = 0;
#pragma unroll 1
  for (; k0 < 1024 - 64; k0 += 64) {
    synth(1, k0 + 32);       // DS ops in-order: stores to buf1 precede next reads
    compute(0, k0);
    __syncthreads();
    synth(0, k0 + 64);
    compute(1, k0 + 32);
    __syncthreads();
  }
  synth(1, k0 + 32);
  compute(0, k0);
  __syncthreads();
  compute(1, k0 + 32);

#pragma unroll
  for (int ms = 0; ms < 2; ++ms)
#pragma unroll
    for (int ns = 0; ns < 2; ++ns) {
      const v8f& cc = ms ? (ns ? acc11 : acc10) : (ns ? acc01 : acc00);
      int gc = n0 + wn * 32 + ns * 16 + l16;
#pragma unroll
      for (int r = 0; r < 8; ++r) {
        int gr = m0 + wm * 32 + ms * 16 + g * 8 + r;
        float v = cc[r];
        if (ELU) v = v > 0.f ? v : (__expf(v) - 1.f);
        size_t idx = ((size_t)(b << 10) + gr) * FO + gc;
        if (OUT16) C16[idx] = (half_t)v;
        else       C32[idx] = v;
      }
    }
}

// ---------------------------------------------------------------------------
__global__ __launch_bounds__(256) void rowdot2(const half_t* __restrict__ H,
                                               const float* __restrict__ a,
                                               float* __restrict__ el,
                                               float* __restrict__ er, int F) {
  int row  = (blockIdx.x * 256 + threadIdx.x) >> 5;
  int lane = threadIdx.x & 31;
  const half_t* hr = H + (size_t)row * F;
  float s0 = 0.f, s1 = 0.f;
  for (int f = lane; f < F; f += 32) {
    float v = (float)hr[f];
    s0 += v * a[f];
    s1 += v * a[F + f];
  }
  for (int o = 16; o; o >>= 1) { s0 += __shfl_down(s0, o); s1 += __shfl_down(s1, o); }
  if (lane == 0) { el[row] = s0; er[row] = s1; }
}

// e = leaky(el_i + er_j) is monotone in (el_i + er_j): NxN min/max collapse
// to vector min/max -> per-batch affine (scale, off) for the sigmoid arg.
__global__ __launch_bounds__(256) void minmax_scale(const float* __restrict__ el,
                                                    const float* __restrict__ er,
                                                    float2* __restrict__ sc) {
  __shared__ float s0[256], s1[256], s2[256], s3[256];
  int b = blockIdx.x, t = threadIdx.x;
  float mnl = 1e30f, mxl = -1e30f, mnr = 1e30f, mxr = -1e30f;
  for (int i = t; i < 1024; i += 256) {
    float v = el[(b << 10) + i]; mnl = fminf(mnl, v); mxl = fmaxf(mxl, v);
    float u = er[(b << 10) + i]; mnr = fminf(mnr, u); mxr = fmaxf(mxr, u);
  }
  s0[t] = mnl; s1[t] = mxl; s2[t] = mnr; s3[t] = mxr;
  __syncthreads();
  for (int o = 128; o; o >>= 1) {
    if (t < o) {
      s0[t] = fminf(s0[t], s0[t + o]); s1[t] = fmaxf(s1[t], s1[t + o]);
      s2[t] = fminf(s2[t], s2[t + o]); s3[t] = fmaxf(s3[t], s3[t + o]);
    }
    __syncthreads();
  }
  if (t == 0) {
    float emn = leaky01(s0[0] + s2[0]);
    float emx = leaky01(s1[0] + s3[0]);
    float scale = 30.f / (emx - emn);
    sc[b] = make_float2(scale, -20.f - emn * scale);
  }
}

__global__ __launch_bounds__(256) void write_att_f32(const float* __restrict__ el,
                                                     const float* __restrict__ er,
                                                     const float2* __restrict__ sc,
                                                     float* __restrict__ out) {
  int bi = blockIdx.x, b = bi >> 10;
  float2 so = sc[b];
  float ei = el[bi];
  const float* erb = er + (b << 10);
  int j0 = threadIdx.x * 4;
  float4 r;
  float* pr = &r.x;
#pragma unroll
  for (int e = 0; e < 4; ++e) {
    float x = leaky01(ei + erb[j0 + e]) * so.x + so.y;
    pr[e] = 1.f / (1.f + __expf(-x));
  }
  *(float4*)(out + (size_t)bi * 1024 + j0) = r;
}

__global__ __launch_bounds__(256) void gemv_g2wg(const float* __restrict__ g2,
                                                 const float* __restrict__ Wg,
                                                 float* __restrict__ gw) {
  int row  = (blockIdx.x * 256 + threadIdx.x) >> 5;
  int lane = threadIdx.x & 31;
  const float* gr = g2 + (size_t)row * 128;
  float s = 0.f;
  for (int f = lane; f < 128; f += 32) s += gr[f] * Wg[f];
  for (int o = 16; o; o >>= 1) s += __shfl_down(s, o);
  if (lane == 0) gw[row] = s;
}

__global__ __launch_bounds__(256) void final_out(const float* __restrict__ el,
                                                 const float* __restrict__ er,
                                                 const float2* __restrict__ sc,
                                                 const float* __restrict__ gw,
                                                 const float* __restrict__ bg,
                                                 float* __restrict__ out) {
  int row  = (blockIdx.x * 256 + threadIdx.x) >> 5;
  int lane = threadIdx.x & 31;
  int b = row >> 10;
  float2 so = sc[b];
  float ei = el[row];
  const float* erb = er + (b << 10);
  const float* gwb = gw + (b << 10);
  float s = 0.f;
  for (int j = lane; j < 1024; j += 32) {
    float x = leaky01(ei + erb[j]) * so.x + so.y;
    s += gwb[j] / (1.f + __expf(-x));
  }
  for (int o = 16; o; o >>= 1) s += __shfl_down(s, o);
  if (lane == 0) {
    float z = s + bg[0];
    out[row] = leaky01(z);
  }
}

// ---------------------------------------------------------------------------
extern "C" void kernel_launch(void* const* d_in, const int* in_sizes, int n_in,
                              void* d_out, int out_size, void* d_ws, size_t ws_size,
                              hipStream_t stream) {
  (void)in_sizes; (void)n_in; (void)out_size; (void)ws_size;
  const float* x  = (const float*)d_in[0];
  // d_in[1] = edge_weight: unused by the reference network
  const float* W1 = (const float*)d_in[2];
  const float* a1 = (const float*)d_in[3];
  const float* W2 = (const float*)d_in[4];
  const float* a2 = (const float*)d_in[5];
  const float* Wg = (const float*)d_in[6];
  const float* bg = (const float*)d_in[7];

  float* out = (float*)d_out;           // [32*1024]
  float* fc2 = out + 32768;             // [32*1024*1024]
  float* g2  = fc2 + 33554432;          // [32*1024*128]

  char* w = (char*)d_ws;
  // region 0: xh (dead after GEMM1) -> reused for Ht1 / h2h / Ht2
  half_t* xh  = (half_t*)(w);                    // 33,554,432 B
  half_t* Ht1 = (half_t*)(w);                    // 16,777,216 B  [32][256][1024]
  half_t* h2h = (half_t*)(w + 16777216);         //  8,388,608 B  [32][1024][128]
  half_t* Ht2 = (half_t*)(w + 25165824);         //  8,388,608 B  [32][128][1024]
  half_t* h1h = (half_t*)(w + 33554432);         // 16,777,216 B  [32][1024][256]
  half_t* g1h = (half_t*)(w + 50331648);         // 16,777,216 B  [32][1024][256]
  char* sm = w + 67108864;
  float*  el1 = (float*)(sm);
  float*  er1 = (float*)(sm + 131072);
  float*  el2 = (float*)(sm + 262144);
  float*  er2 = (float*)(sm + 393216);
  float*  gw  = (float*)(sm + 524288);
  float2* sc1 = (float2*)(sm + 655360);
  float2* sc2 = (float2*)(sm + 655616);
  half_t* W1h = (half_t*)(sm + 1048576);         // [512][256]
  half_t* W1t = (half_t*)(sm + 1310720);         // [256][512]
  half_t* W2h = (half_t*)(sm + 1572864);         // [256][128]
  half_t* W2t = (half_t*)(sm + 1638400);         // [128][256]

  // --- operand conversion + weight transposes ---
  cvt4_f32_f16<<<16384, 256, 0, stream>>>(x,  xh,  16777216 / 4);
  cvt4_f32_f16<<<128,   256, 0, stream>>>(W1, W1h, 131072 / 4);
  cvt4_f32_f16<<<32,    256, 0, stream>>>(W2, W2h, 32768 / 4);
  transpose16<<<dim3(8, 16, 1),  256, 0, stream>>>(W1h, W1t, 512, 256);
  transpose16<<<dim3(4, 8, 1),   256, 0, stream>>>(W2h, W2t, 256, 128);

  // --- layer 1 ---
  gemm_bt<512, 256><<<dim3(2, 512), 256, 0, stream>>>(xh, W1t, h1h);
  rowdot2<<<4096, 256, 0, stream>>>(h1h, a1, el1, er1, 256);
  minmax_scale<<<32, 256, 0, stream>>>(el1, er1, sc1);
  transpose16<<<dim3(8, 32, 32), 256, 0, stream>>>(h1h, Ht1, 1024, 256);
  gat_att_gemm<true, true, 256><<<dim3(2, 16, 32), 256, 0, stream>>>(el1, er1, sc1, Ht1,
                                                                     g1h, nullptr);
  // --- layer 2 ---
  gemm_bt<256, 128><<<dim3(1, 512), 256, 0, stream>>>(g1h, W2t, h2h);
  rowdot2<<<4096, 256, 0, stream>>>(h2h, a2, el2, er2, 128);
  minmax_scale<<<32, 256, 0, stream>>>(el2, er2, sc2);
  write_att_f32<<<32768, 256, 0, stream>>>(el2, er2, sc2, fc2);
  transpose16<<<dim3(4, 32, 32), 256, 0, stream>>>(h2h, Ht2, 1024, 128);
  gat_att_gemm<false, false, 128><<<dim3(1, 16, 32), 256, 0, stream>>>(el2, er2, sc2, Ht2,
                                                                       nullptr, g2);
  // --- head: (fc2 @ g2) @ Wg + bg == fc2 @ (g2 @ Wg) + bg ---
  gemv_g2wg<<<4096, 256, 0, stream>>>(g2, Wg, gw);
  final_out<<<4096, 256, 0, stream>>>(el2, er2, sc2, gw, bg, out);
}
